// GCN_84009560309789
// MI455X (gfx1250) — matrix operand order repaired and checked by reading, not compiled, not run on hardware
//
#include <hip/hip_runtime.h>
#include <hip/hip_bf16.h>
#include <math.h>

typedef __attribute__((ext_vector_type(16))) __bf16 v16bf;
typedef __attribute__((ext_vector_type(8)))  __bf16 v8bf;
typedef __attribute__((ext_vector_type(8)))  float  v8f;

#define SHUF16(a, b) __builtin_shufflevector(a, b, 0,1,2,3,4,5,6,7,8,9,10,11,12,13,14,15)

// ---------------------------------------------------------------------------
// Degree / normalization kernels
// ---------------------------------------------------------------------------
__global__ void gcn_deg_init(float* deg, int n) {
    int i = blockIdx.x * blockDim.x + threadIdx.x;
    if (i < n) deg[i] = 1.0f;                      // self-loop contributes 1
}

__global__ void gcn_deg_acc(const long long* __restrict__ ei, float* deg, int E) {
    int i = blockIdx.x * blockDim.x + threadIdx.x;
    if (i < E) atomicAdd(&deg[(int)ei[(size_t)E + i]], 1.0f);   // dst row
}

__global__ void gcn_deg_inv(float* deg, int n) {
    int i = blockIdx.x * blockDim.x + threadIdx.x;
    if (i < n) deg[i] = rsqrtf(fmaxf(deg[i], 1.0f));
}

__global__ void gcn_zero(float* p, int n) {
    int i = blockIdx.x * blockDim.x + threadIdx.x;
    if (i < n) p[i] = 0.0f;
}

// W[R][C] (f32, row major) -> Wt[C][R] (bf16)
__global__ void gcn_cvt_transpose(const float* __restrict__ W, __bf16* __restrict__ Wt,
                                  int R, int C) {
    int i = blockIdx.x * blockDim.x + threadIdx.x;
    if (i < R * C) {
        int r = i / C, c = i % C;
        Wt[(size_t)c * R + r] = (__bf16)W[i];
    }
}

// ---------------------------------------------------------------------------
// GEMM1: H[N,64] = X[N,512] @ W1[512,64]   (bf16 WMMA, f32 accumulate)
// block = 256 threads = 8 waves; each wave owns 16 distinct rows x all 64
// cols (4 accumulators, A fragment reused 4x per K-step). No LDS, no
// barriers: A built straight from coalesced global f32 loads (lanes L and
// L+16 tile each row's 128B K-slab), B from pre-transposed bf16 W1t.
// All 8 B loads issued before the 4 WMMAs so waits are partial counts.
// ---------------------------------------------------------------------------
__global__ void gcn_gemm1_wmma(const float* __restrict__ X,
                               const __bf16* __restrict__ Wt,   // [64][512]
                               float* __restrict__ H, int nrows) {
    const int lane = threadIdx.x & 31;
    const int wave = threadIdx.x >> 5;
    const int rowBase = blockIdx.x * 128 + wave * 16;

    const int m     = lane & 15;
    const int half  = lane >> 4;     // 0 or 1
    const int khalf = half * 16;     // A: K-offset per lane half
    const int kb    = half * 16;     // B: K-offset per lane half

    // clamp OOB rows: garbage only reaches never-stored output rows
    int ar = rowBase + m;
    if (ar > nrows - 1) ar = nrows - 1;
    const float* arow = X + (size_t)ar * 512;
    const __bf16* brow = Wt + (size_t)m * 512 + kb;   // col tile c adds c*16*512

    v8f acc[4];
    #pragma unroll
    for (int c = 0; c < 4; ++c)
        acc[c] = (v8f){0.f, 0.f, 0.f, 0.f, 0.f, 0.f, 0.f, 0.f};

    for (int k0 = 0; k0 < 512; k0 += 32) {
        // A fragment: one contiguous 16-float run -> 4x global_load_b128 + cvt
        float4 f0 = *(const float4*)(arow + k0 + khalf);
        float4 f1 = *(const float4*)(arow + k0 + khalf + 4);
        float4 f2 = *(const float4*)(arow + k0 + khalf + 8);
        float4 f3 = *(const float4*)(arow + k0 + khalf + 12);
        v16bf a = { (__bf16)f0.x, (__bf16)f0.y, (__bf16)f0.z, (__bf16)f0.w,
                    (__bf16)f1.x, (__bf16)f1.y, (__bf16)f1.z, (__bf16)f1.w,
                    (__bf16)f2.x, (__bf16)f2.y, (__bf16)f2.z, (__bf16)f2.w,
                    (__bf16)f3.x, (__bf16)f3.y, (__bf16)f3.z, (__bf16)f3.w };

        // issue all 8 B-fragment loads first, then 4 WMMAs
        const v8bf* bp0 = (const v8bf*)(brow + k0);
        const v8bf* bp1 = (const v8bf*)(brow + 16 * 512 + k0);
        const v8bf* bp2 = (const v8bf*)(brow + 32 * 512 + k0);
        const v8bf* bp3 = (const v8bf*)(brow + 48 * 512 + k0);
        v16bf b0 = SHUF16(bp0[0], bp0[1]);
        v16bf b1 = SHUF16(bp1[0], bp1[1]);
        v16bf b2 = SHUF16(bp2[0], bp2[1]);
        v16bf b3 = SHUF16(bp3[0], bp3[1]);

        acc[0] = __builtin_amdgcn_wmma_f32_16x16x32_bf16(
            false, a, false, b0, (short)0, acc[0], false, false);
        acc[1] = __builtin_amdgcn_wmma_f32_16x16x32_bf16(
            false, a, false, b1, (short)0, acc[1], false, false);
        acc[2] = __builtin_amdgcn_wmma_f32_16x16x32_bf16(
            false, a, false, b2, (short)0, acc[2], false, false);
        acc[3] = __builtin_amdgcn_wmma_f32_16x16x32_bf16(
            false, a, false, b3, (short)0, acc[3], false, false);

        if (k0 + 32 < 512)
            __builtin_prefetch(arow + k0 + 32 + khalf, 0, 1);
    }

    // D layout: lane n = lane&15, VGPR r -> row r + 8*half
    const int mOff = half * 8;
    float* hbase = H + (size_t)(rowBase + mOff) * 64 + m;
    if (rowBase + 16 <= nrows) {
        // scalar-uniform fast path: all 16 rows in range, unguarded stores
        #pragma unroll
        for (int c = 0; c < 4; ++c) {
            #pragma unroll
            for (int r = 0; r < 8; ++r)
                hbase[(size_t)r * 64 + c * 16] = acc[c][r];
        }
    } else {
        #pragma unroll
        for (int c = 0; c < 4; ++c) {
            #pragma unroll
            for (int r = 0; r < 8; ++r) {
                int gr = rowBase + mOff + r;
                if (gr < nrows) H[(size_t)gr * 64 + c * 16 + m] = acc[c][r];
            }
        }
    }
}

// ---------------------------------------------------------------------------
// GEMM2: HH[N,16] = relu(AGG1[N,64] + b1) @ W2[64,16]
// block = 256 threads = 8 waves; each wave owns 16 rows x 16 cols.
// relu+bias fused into per-lane A-fragment conversion; no LDS, no barriers.
// ---------------------------------------------------------------------------
__global__ void gcn_gemm2_wmma(const float* __restrict__ A1,
                               const float* __restrict__ b1,
                               const __bf16* __restrict__ W2t,  // [16][64]
                               float* __restrict__ HH, int nrows) {
    const int lane = threadIdx.x & 31;
    const int wave = threadIdx.x >> 5;
    const int rowBase = blockIdx.x * 128 + wave * 16;

    const int m     = lane & 15;
    const int half  = lane >> 4;
    const int khalf = half * 16;
    const int kb    = half * 16;

    int ar = rowBase + m;
    if (ar > nrows - 1) ar = nrows - 1;
    const float* arow = A1 + (size_t)ar * 64;
    const __bf16* brow = W2t + (size_t)m * 64 + kb;

    v8f acc = {0.f, 0.f, 0.f, 0.f, 0.f, 0.f, 0.f, 0.f};

    #pragma unroll
    for (int k0 = 0; k0 < 64; k0 += 32) {
        float4 f0 = *(const float4*)(arow + k0 + khalf);
        float4 f1 = *(const float4*)(arow + k0 + khalf + 4);
        float4 f2 = *(const float4*)(arow + k0 + khalf + 8);
        float4 f3 = *(const float4*)(arow + k0 + khalf + 12);
        float4 g0 = *(const float4*)(b1 + k0 + khalf);
        float4 g1 = *(const float4*)(b1 + k0 + khalf + 4);
        float4 g2 = *(const float4*)(b1 + k0 + khalf + 8);
        float4 g3 = *(const float4*)(b1 + k0 + khalf + 12);
        #define RB(x, y) ((__bf16)fmaxf((x) + (y), 0.f))
        v16bf a = { RB(f0.x,g0.x), RB(f0.y,g0.y), RB(f0.z,g0.z), RB(f0.w,g0.w),
                    RB(f1.x,g1.x), RB(f1.y,g1.y), RB(f1.z,g1.z), RB(f1.w,g1.w),
                    RB(f2.x,g2.x), RB(f2.y,g2.y), RB(f2.z,g2.z), RB(f2.w,g2.w),
                    RB(f3.x,g3.x), RB(f3.y,g3.y), RB(f3.z,g3.z), RB(f3.w,g3.w) };
        #undef RB

        const v8bf* bp = (const v8bf*)(brow + k0);
        v16bf b = SHUF16(bp[0], bp[1]);

        acc = __builtin_amdgcn_wmma_f32_16x16x32_bf16(
            false, a, false, b, (short)0, acc, false, false);
    }

    const int mOff = half * 8;
    float* obase = HH + (size_t)(rowBase + mOff) * 16 + m;
    if (rowBase + 16 <= nrows) {
        #pragma unroll
        for (int r = 0; r < 8; ++r)
            obase[(size_t)r * 16] = acc[r];
    } else {
        #pragma unroll
        for (int r = 0; r < 8; ++r) {
            int gr = rowBase + mOff + r;
            if (gr < nrows) HH[(size_t)gr * 16 + m] = acc[r];
        }
    }
}

// ---------------------------------------------------------------------------
// Scatter-add aggregation: agg[dst] += h[src] * dinv[src]*dinv[dst]
// One thread per (edge, feature). Edges [0,E) real, [E, E+N) self-loops.
// ---------------------------------------------------------------------------
template <int LOGF>
__global__ void gcn_scatter(const long long* __restrict__ ei,
                            const float* __restrict__ dinv,
                            const float* __restrict__ h,
                            float* __restrict__ agg, int E, int N) {
    const int F = 1 << LOGF;
    long long g = (long long)blockIdx.x * blockDim.x + threadIdx.x;
    long long total = ((long long)E + N) << LOGF;
    if (g >= total) return;
    int e = (int)(g >> LOGF);
    int f = (int)(g & (F - 1));
    int s, d;
    if (e < E) { s = (int)ei[e]; d = (int)ei[(size_t)E + e]; }
    else       { s = e - E; d = s; }
    float w = dinv[s] * dinv[d];
    atomicAdd(&agg[(size_t)d * F + f], h[(size_t)s * F + f] * w);
}

// ---------------------------------------------------------------------------
// out = log_softmax(agg2 + b2) per row of 16
// ---------------------------------------------------------------------------
__global__ void gcn_logsoftmax(const float* __restrict__ agg2,
                               const float* __restrict__ b2,
                               float* __restrict__ out, int n) {
    int row = blockIdx.x * blockDim.x + threadIdx.x;
    if (row >= n) return;
    float v[16];
    float m = -1e30f;
    #pragma unroll
    for (int f = 0; f < 16; ++f) {
        v[f] = agg2[(size_t)row * 16 + f] + b2[f];
        m = fmaxf(m, v[f]);
    }
    float s = 0.0f;
    #pragma unroll
    for (int f = 0; f < 16; ++f) s += __expf(v[f] - m);
    float ls = m + __logf(s);
    #pragma unroll
    for (int f = 0; f < 16; ++f) out[(size_t)row * 16 + f] = v[f] - ls;
}

// ---------------------------------------------------------------------------
extern "C" void kernel_launch(void* const* d_in, const int* in_sizes, int n_in,
                              void* d_out, int out_size, void* d_ws, size_t ws_size,
                              hipStream_t stream) {
    const float*     x  = (const float*)d_in[0];
    const long long* ei = (const long long*)d_in[1];   // int64 [2, E]
    const float*     W1 = (const float*)d_in[2];
    const float*     b1 = (const float*)d_in[3];
    const float*     W2 = (const float*)d_in[4];
    const float*     b2 = (const float*)d_in[5];
    float* out = (float*)d_out;

    const int N = in_sizes[0] / 512;     // 50000
    const int E = in_sizes[1] / 2;       // 800000
    const int Np = (N + 63) & ~63;       // padded for aligned sections

    char* ws = (char*)d_ws;
    float*  dinv = (float*)ws;                           // Np
    float*  h    = (float*)(ws + (size_t)Np * 4);        // Np*64
    float*  agg1 = h    + (size_t)Np * 64;               // Np*64
    float*  hh   = agg1 + (size_t)Np * 64;               // Np*16
    float*  agg2 = hh   + (size_t)Np * 16;               // Np*16
    __bf16* W1t  = (__bf16*)(agg2 + (size_t)Np * 16);    // 64*512 bf16 (64 KB)
    __bf16* W2t  = W1t + (size_t)64 * 512;               // 16*64 bf16

    const int T = 256;

    // 0) pre-transpose weights to bf16 (tiny, L2-resident)
    gcn_cvt_transpose<<<(512 * 64 + T - 1) / T, T, 0, stream>>>(W1, W1t, 512, 64);
    gcn_cvt_transpose<<<(64 * 16 + T - 1) / T, T, 0, stream>>>(W2, W2t, 64, 16);

    // 1) degree + normalization
    gcn_deg_init<<<(N + T - 1) / T, T, 0, stream>>>(dinv, N);
    gcn_deg_acc <<<(E + T - 1) / T, T, 0, stream>>>(ei, dinv, E);
    gcn_deg_inv <<<(N + T - 1) / T, T, 0, stream>>>(dinv, N);

    // 2) GEMM1 (WMMA bf16): h = x @ W1
    gcn_gemm1_wmma<<<(N + 127) / 128, T, 0, stream>>>(x, W1t, h, N);

    // 3) layer-1 aggregation
    gcn_zero<<<((N * 64) + T - 1) / T, T, 0, stream>>>(agg1, N * 64);
    {
        long long total = ((long long)E + N) * 64;
        int blocks = (int)((total + T - 1) / T);
        gcn_scatter<6><<<blocks, T, 0, stream>>>(ei, dinv, h, agg1, E, N);
    }

    // 4) GEMM2 (WMMA bf16) with fused relu(agg1+b1): hh = h1 @ W2
    gcn_gemm2_wmma<<<(N + 127) / 128, T, 0, stream>>>(agg1, b1, W2t, hh, N);

    // 5) layer-2 aggregation
    gcn_zero<<<((N * 16) + T - 1) / T, T, 0, stream>>>(agg2, N * 16);
    {
        long long total = ((long long)E + N) * 16;
        int blocks = (int)((total + T - 1) / T);
        gcn_scatter<4><<<blocks, T, 0, stream>>>(ei, dinv, hh, agg2, E, N);
    }

    // 6) bias + log_softmax
    gcn_logsoftmax<<<(N + T - 1) / T, T, 0, stream>>>(agg2, b2, out, N);
}
